// Pointnet2SSG_54606214201518
// MI455X (gfx1250) — matrix operand order repaired and checked by reading, not compiled, not run on hardware
//
#include <hip/hip_runtime.h>
#include <hip/hip_bf16.h>

// ---------------------------------------------------------------------------
// PointNet++ SSG forward for MI455X (gfx1250, wave32).
// All shared-MLP matmuls go through v_wmma_f32_16x16x32_f16.
// Weights are pre-packed per launch into B-fragment order (zero-padded f16)
// so every B fragment is one contiguous 32-byte load per lane.
// ---------------------------------------------------------------------------

typedef _Float16 v16h __attribute__((ext_vector_type(16)));
typedef _Float16 v8h  __attribute__((ext_vector_type(8)));
typedef float    v8f  __attribute__((ext_vector_type(8)));

#define BATCH 32
#define NPTS  4096

// ---- WMMA fragment loaders (A and B share a contiguous K order) ----

// A fragment 16x32 (MxK), this code's K order: lanes 0-15 -> M=lane, K in {0..15};
// lanes 16-31 -> M=lane-16, K in {16..31}. Each lane reads a contiguous
// 16-half run -> two 128-bit loads.
static __device__ __forceinline__ v16h load_a_frag(const _Float16* act, int stride,
                                                   int mbase, int k0, int lane) {
  const _Float16* p = act + (size_t)(mbase + (lane & 15)) * stride + k0
                          + ((lane < 16) ? 0 : 16);
  v8h lo = *(const v8h*)p;
  v8h hi = *(const v8h*)(p + 8);
  return __builtin_shufflevector(lo, hi, 0, 1, 2, 3, 4, 5, 6, 7,
                                 8, 9, 10, 11, 12, 13, 14, 15);
}

// Packed-weight B fragment: P[(ktile*Cout + n)*32 + j] holds f16 of
// W[ktile*32 + j][n] (zero-padded past Cin). Lane l (N = n0 + l&15) reads 16
// contiguous halves starting at j = (l<16 ? 0 : 16)  ->  K = k0+{0..15|16..31}.
static __device__ __forceinline__ v16h load_b_frag(const _Float16* P, int Cout,
                                                   int k0, int n0, int lane) {
  const _Float16* p = P + ((size_t)(k0 >> 5) * Cout + (n0 + (lane & 15))) * 32
                        + ((lane < 16) ? 0 : 16);
  return *(const v16h*)p;
}

// One MLP layer for a 64-row group held in LDS. 4 waves; wave w owns rows
// [16w,16w+16). N processed in groups of 4 tiles (Cout % 64 == 0) so each A
// fragment feeds 4 back-to-back WMMAs.
// C/D layout: VGPR r, lane l -> M = r + (l>=16 ? 8:0), N = l&15.
static __device__ __forceinline__ void mlp_layer(const _Float16* actIn, int strideIn,
                                                 _Float16* actOut, int strideOut,
                                                 const _Float16* Wp, const float* gamma,
                                                 const float* beta, int Cin, int Cout,
                                                 int lane, int wave) {
  int mbase = wave * 16;
  int Kpad  = (Cin + 31) & ~31;
  for (int n0g = 0; n0g < Cout; n0g += 64) {
    v8f acc[4] = {{}, {}, {}, {}};
    for (int k0 = 0; k0 < Kpad; k0 += 32) {
      v16h a = load_a_frag(actIn, strideIn, mbase, k0, lane);
      if (k0 + 32 < Kpad)
        __builtin_prefetch(Wp + ((size_t)((k0 >> 5) + 1) * Cout + n0g) * 32, 0, 0);
#pragma unroll
      for (int nt = 0; nt < 4; ++nt) {
        v16h b = load_b_frag(Wp, Cout, k0, n0g + nt * 16, lane);
        acc[nt] = __builtin_amdgcn_wmma_f32_16x16x32_f16(false, a, false, b,
                                                         (short)0, acc[nt], false, false);
      }
    }
    int moff = mbase + ((lane < 16) ? 0 : 8);
#pragma unroll
    for (int nt = 0; nt < 4; ++nt) {
      int n   = n0g + nt * 16 + (lane & 15);
      float g = gamma[n], bt = beta[n];
#pragma unroll
      for (int r = 0; r < 8; ++r) {
        float v = acc[nt][r] * g + bt;
        v = v > 0.0f ? v : 0.0f;
        actOut[(size_t)(moff + r) * strideOut + n] = (_Float16)v;
      }
    }
  }
}

// ---------------------------------------------------------------------------
// Weight pre-pack: f32 (Cin x Cout) -> f16 fragment-ordered, zero-padded to
// Kpad = ceil(Cin/32)*32.  P[(ktile*Cout + n)*32 + j] = W[ktile*32 + j][n].
// ---------------------------------------------------------------------------
__global__ void pack_w_kernel(const float* __restrict__ W, int Cin, int Cout,
                              _Float16* __restrict__ P) {
  int t = blockIdx.x * blockDim.x + threadIdx.x;
  int Kpad = (Cin + 31) & ~31;
  int total = Kpad * Cout;
  if (t >= total) return;
  int ktile = t / (Cout * 32);
  int rem   = t - ktile * Cout * 32;
  int n     = rem >> 5;
  int j     = rem & 31;
  int k     = ktile * 32 + j;
  P[t] = (_Float16)((k < Cin) ? W[(size_t)k * Cout + n] : 0.0f);
}

// ---------------------------------------------------------------------------
// Farthest point sampling: one block per batch, dist[] in LDS, argmax with
// smallest-index tie-break (matches jnp.argmax). Writes sampled coords.
// ---------------------------------------------------------------------------
__global__ void fps_kernel(const float* __restrict__ xyz, int n, int npoint,
                           float* __restrict__ new_xyz) {
  __shared__ float dist[NPTS];
  __shared__ float rv[256];
  __shared__ int   ri[256];
  __shared__ int   last_s;
  int b = blockIdx.x, tid = threadIdx.x;
  const float* pts = xyz + (size_t)b * n * 3;
  for (int j = tid; j < n; j += 256) dist[j] = 1e10f;
  if (tid == 0) {
    last_s = 0;
    new_xyz[(size_t)(b * npoint) * 3 + 0] = pts[0];
    new_xyz[(size_t)(b * npoint) * 3 + 1] = pts[1];
    new_xyz[(size_t)(b * npoint) * 3 + 2] = pts[2];
  }
  __syncthreads();
  for (int it = 1; it < npoint; ++it) {
    int last = last_s;
    float px = pts[last * 3], py = pts[last * 3 + 1], pz = pts[last * 3 + 2];
    float bv = -1.0f; int bi = n - 1;
    for (int j = tid; j < n; j += 256) {
      float dx = pts[j * 3] - px, dy = pts[j * 3 + 1] - py, dz = pts[j * 3 + 2] - pz;
      float d  = dx * dx + dy * dy + dz * dz;
      float od = dist[j];
      float nd = d < od ? d : od;
      dist[j]  = nd;
      if (nd > bv) { bv = nd; bi = j; }
    }
    rv[tid] = bv; ri[tid] = bi;
    __syncthreads();
    for (int s = 128; s > 0; s >>= 1) {
      if (tid < s) {
        float ov = rv[tid + s]; int oi = ri[tid + s];
        if (ov > rv[tid] || (ov == rv[tid] && oi < ri[tid])) { rv[tid] = ov; ri[tid] = oi; }
      }
      __syncthreads();
    }
    if (tid == 0) {
      int sel = ri[0];
      last_s  = sel;
      new_xyz[(size_t)(b * npoint + it) * 3 + 0] = pts[sel * 3];
      new_xyz[(size_t)(b * npoint + it) * 3 + 1] = pts[sel * 3 + 1];
      new_xyz[(size_t)(b * npoint + it) * 3 + 2] = pts[sel * 3 + 2];
    }
    __syncthreads();
  }
}

// ---------------------------------------------------------------------------
// Ball query: one thread per center; ordered scan = first nsample in-ball
// indices in increasing order (== reference's sort-of-keyed-index trick),
// deficits padded with the first hit.
// ---------------------------------------------------------------------------
__global__ void ballquery_kernel(const float* __restrict__ xyz, int n,
                                 const float* __restrict__ centers, int total,
                                 float r2, int* __restrict__ idx) {
  int t = blockIdx.x * blockDim.x + threadIdx.x;
  if (t >= total) return;
  int S = total / BATCH;
  int b = t / S;
  const float* pts = xyz + (size_t)b * n * 3;
  float cx = centers[(size_t)t * 3], cy = centers[(size_t)t * 3 + 1], cz = centers[(size_t)t * 3 + 2];
  int* out = idx + (size_t)t * 64;
  int cnt = 0, first = n - 1;
  for (int j = 0; j < n && cnt < 64; ++j) {
    float dx = pts[j * 3] - cx, dy = pts[j * 3 + 1] - cy, dz = pts[j * 3 + 2] - cz;
    if (dx * dx + dy * dy + dz * dz < r2) {
      if (cnt == 0) first = j;
      out[cnt++] = j;
    }
  }
  if (cnt == 0) { out[0] = n - 1; cnt = 1; }
  for (; cnt < 64; ++cnt) out[cnt] = first;
}

// ---------------------------------------------------------------------------
// SA1 fused: gather relative xyz -> LDS f16 -> 3 WMMA MLP layers -> maxpool.
// One workgroup (4 waves) per center. LDS = 2 * 64*128 f16 = 32 KB.
// ---------------------------------------------------------------------------
__global__ __launch_bounds__(128) void sa1_kernel(
    const float* __restrict__ xyz, const int* __restrict__ idx,
    const float* __restrict__ centers,
    const _Float16* w0, const float* g0, const float* b0,
    const _Float16* w1, const float* g1, const float* b1,
    const _Float16* w2, const float* g2, const float* b2,
    float* __restrict__ feat_out) {
  __shared__ _Float16 bufA[64 * 128];
  __shared__ _Float16 bufB[64 * 128];
  int bs = blockIdx.x;                    // b*512 + s
  int tid = threadIdx.x, lane = tid & 31, wave = tid >> 5;
  for (int i = tid; i < 64 * 128; i += 128) { bufA[i] = (_Float16)0.0f; bufB[i] = (_Float16)0.0f; }
  __syncthreads();
  int b = bs >> 9;
  const float* pts = xyz + (size_t)b * NPTS * 3;
  const int* gidx  = idx + (size_t)bs * 64;
  float cx = centers[(size_t)bs * 3], cy = centers[(size_t)bs * 3 + 1], cz = centers[(size_t)bs * 3 + 2];
  for (int e = tid; e < 64 * 3; e += 128) {
    int j = e / 3, c = e - j * 3;
    int p = gidx[j];
    float v = pts[p * 3 + c] - (c == 0 ? cx : (c == 1 ? cy : cz));
    bufA[j * 128 + c] = (_Float16)v;
  }
  __syncthreads();
  mlp_layer(bufA, 128, bufB, 128, w0, g0, b0, 3, 64, lane, wave);
  __syncthreads();
  mlp_layer(bufB, 128, bufA, 128, w1, g1, b1, 64, 64, lane, wave);
  __syncthreads();
  mlp_layer(bufA, 128, bufB, 128, w2, g2, b2, 64, 128, lane, wave);
  __syncthreads();
  for (int c = tid; c < 128; c += 128) {
    float m = 0.0f;  // post-ReLU values are >= 0
    for (int j = 0; j < 64; ++j) { float v = (float)bufB[j * 128 + c]; m = v > m ? v : m; }
    feat_out[(size_t)bs * 128 + c] = m;
  }
}

// ---------------------------------------------------------------------------
// SA2 fused: gather [dxyz | feat1] (131ch, padded to 160) -> 3 WMMA layers ->
// maxpool 256ch. LDS = 64*160 + 64*256 f16 = 52 KB.
// ---------------------------------------------------------------------------
__global__ __launch_bounds__(128) void sa2_kernel(
    const float* __restrict__ xyz1, const int* __restrict__ idx,
    const float* __restrict__ centers, const float* __restrict__ feat1,
    const _Float16* w0, const float* g0, const float* b0,
    const _Float16* w1, const float* g1, const float* b1,
    const _Float16* w2, const float* g2, const float* b2,
    float* __restrict__ feat_out) {
  __shared__ _Float16 bufA[64 * 160];
  __shared__ _Float16 bufB[64 * 256];
  int bs = blockIdx.x;                    // b*128 + s
  int tid = threadIdx.x, lane = tid & 31, wave = tid >> 5;
  for (int i = tid; i < 64 * 160; i += 128) bufA[i] = (_Float16)0.0f;
  for (int i = tid; i < 64 * 256; i += 128) bufB[i] = (_Float16)0.0f;
  __syncthreads();
  int b = bs >> 7;
  const float* pts = xyz1 + (size_t)b * 512 * 3;
  const int* gidx  = idx + (size_t)bs * 64;
  float cx = centers[(size_t)bs * 3], cy = centers[(size_t)bs * 3 + 1], cz = centers[(size_t)bs * 3 + 2];
  for (int e = tid; e < 64 * 131; e += 128) {
    int j = e / 131, c = e - j * 131;
    int p = gidx[j];
    float v;
    if (c < 3) v = pts[p * 3 + c] - (c == 0 ? cx : (c == 1 ? cy : cz));
    else       v = feat1[((size_t)b * 512 + p) * 128 + (c - 3)];
    bufA[j * 160 + c] = (_Float16)v;
  }
  __syncthreads();
  mlp_layer(bufA, 160, bufB, 256, w0, g0, b0, 131, 128, lane, wave);
  __syncthreads();
  mlp_layer(bufB, 256, bufA, 160, w1, g1, b1, 128, 128, lane, wave);
  __syncthreads();
  mlp_layer(bufA, 160, bufB, 256, w2, g2, b2, 128, 256, lane, wave);
  __syncthreads();
  for (int c = tid; c < 256; c += 128) {
    float m = 0.0f;
    for (int j = 0; j < 64; ++j) { float v = (float)bufB[j * 256 + c]; m = v > m ? v : m; }
    feat_out[(size_t)bs * 256 + c] = m;
  }
}

// ---------------------------------------------------------------------------
// Build SA3 input A matrix: (B*128) x 288 f16, [xyz2 | feat2 | zero-pad].
// ---------------------------------------------------------------------------
__global__ void build_a3_kernel(const float* __restrict__ xyz2,
                                const float* __restrict__ feat2,
                                _Float16* __restrict__ a3) {
  int t = blockIdx.x * blockDim.x + threadIdx.x;
  const int total = BATCH * 128 * 288;
  if (t >= total) return;
  int row = t / 288, c = t - row * 288;
  float v = 0.0f;
  if (c < 3)        v = xyz2[(size_t)row * 3 + c];
  else if (c < 259) v = feat2[(size_t)row * 256 + (c - 3)];
  a3[t] = (_Float16)v;
}

// ---------------------------------------------------------------------------
// Generic WMMA GEMM + folded-BN + ReLU for SA3. Block = 4 waves computes a
// 64x64 tile; wave w owns M-tile w. Writes f16 (intermediate) or f32 (last).
// ---------------------------------------------------------------------------
__global__ __launch_bounds__(128) void gemm_relu_kernel(
    const _Float16* __restrict__ A, int strideA,
    const _Float16* __restrict__ Wp, const float* __restrict__ gamma,
    const float* __restrict__ beta, int Cin, int Cout,
    _Float16* __restrict__ out16, float* __restrict__ out32, int strideOut) {
  int lane = threadIdx.x & 31, wave = threadIdx.x >> 5;
  int mbase = blockIdx.y * 64 + wave * 16;
  int nbase = blockIdx.x * 64;
  int Kpad  = (Cin + 31) & ~31;
  v8f acc[4] = {{}, {}, {}, {}};
  for (int k0 = 0; k0 < Kpad; k0 += 32) {
    v16h a = load_a_frag(A, strideA, mbase, k0, lane);
    if (k0 + 32 < Kpad)
      __builtin_prefetch(Wp + ((size_t)((k0 >> 5) + 1) * Cout + nbase) * 32, 0, 0);
#pragma unroll
    for (int nt = 0; nt < 4; ++nt) {
      v16h b = load_b_frag(Wp, Cout, k0, nbase + nt * 16, lane);
      acc[nt] = __builtin_amdgcn_wmma_f32_16x16x32_f16(false, a, false, b,
                                                       (short)0, acc[nt], false, false);
    }
  }
  int moff = mbase + ((lane < 16) ? 0 : 8);
  if (out16) {
#pragma unroll
    for (int nt = 0; nt < 4; ++nt) {
      int n   = nbase + nt * 16 + (lane & 15);
      float g = gamma[n], bt = beta[n];
#pragma unroll
      for (int r = 0; r < 8; ++r) {
        float v = acc[nt][r] * g + bt;
        v = v > 0.0f ? v : 0.0f;
        out16[(size_t)(moff + r) * strideOut + n] = (_Float16)v;
      }
    }
  } else {
#pragma unroll
    for (int nt = 0; nt < 4; ++nt) {
      int n   = nbase + nt * 16 + (lane & 15);
      float g = gamma[n], bt = beta[n];
#pragma unroll
      for (int r = 0; r < 8; ++r) {
        float v = acc[nt][r] * g + bt;
        v = v > 0.0f ? v : 0.0f;
        out32[(size_t)(moff + r) * strideOut + n] = v;
      }
    }
  }
}

// Final max over the 128 points per batch -> (B, 1024) f32.
__global__ void max3_kernel(const float* __restrict__ h, float* __restrict__ out) {
  int b = blockIdx.x;
  int c = blockIdx.y * 256 + threadIdx.x;
  float m = 0.0f;
  for (int s = 0; s < 128; ++s) {
    float v = h[((size_t)b * 128 + s) * 1024 + c];
    m = v > m ? v : m;
  }
  out[(size_t)b * 1024 + c] = m;
}

// ---------------------------------------------------------------------------
extern "C" void kernel_launch(void* const* d_in, const int* in_sizes, int n_in,
                              void* d_out, int out_size, void* d_ws, size_t ws_size,
                              hipStream_t stream) {
  const float* pc  = (const float*)d_in[0];
  const float* w10 = (const float*)d_in[1],  *g10 = (const float*)d_in[2],  *b10 = (const float*)d_in[3];
  const float* w11 = (const float*)d_in[4],  *g11 = (const float*)d_in[5],  *b11 = (const float*)d_in[6];
  const float* w12 = (const float*)d_in[7],  *g12 = (const float*)d_in[8],  *b12 = (const float*)d_in[9];
  const float* w20 = (const float*)d_in[10], *g20 = (const float*)d_in[11], *b20 = (const float*)d_in[12];
  const float* w21 = (const float*)d_in[13], *g21 = (const float*)d_in[14], *b21 = (const float*)d_in[15];
  const float* w22 = (const float*)d_in[16], *g22 = (const float*)d_in[17], *b22 = (const float*)d_in[18];
  const float* w30 = (const float*)d_in[19], *g30 = (const float*)d_in[20], *b30 = (const float*)d_in[21];
  const float* w31 = (const float*)d_in[22], *g31 = (const float*)d_in[23], *b31 = (const float*)d_in[24];
  const float* w32 = (const float*)d_in[25], *g32 = (const float*)d_in[26], *b32 = (const float*)d_in[27];

  char* ws = (char*)d_ws;
  size_t off = 0;
  auto take = [&](size_t bytes) { char* p = ws + off; off += (bytes + 255) & ~(size_t)255; return p; };
  float*    xyz1  = (float*)   take((size_t)BATCH * 512 * 3 * 4);
  int*      idx1  = (int*)     take((size_t)BATCH * 512 * 64 * 4);
  float*    feat1 = (float*)   take((size_t)BATCH * 512 * 128 * 4);
  float*    xyz2  = (float*)   take((size_t)BATCH * 128 * 3 * 4);
  int*      idx2  = (int*)     take((size_t)BATCH * 128 * 64 * 4);
  float*    feat2 = (float*)   take((size_t)BATCH * 128 * 256 * 4);
  _Float16* a3    = (_Float16*)take((size_t)BATCH * 128 * 288 * 2);
  _Float16* h31   = (_Float16*)take((size_t)BATCH * 128 * 256 * 2);
  _Float16* h32   = (_Float16*)take((size_t)BATCH * 128 * 512 * 2);
  float*    h33   = (float*)   take((size_t)BATCH * 128 * 1024 * 4);
  // packed f16 weights (Kpad * Cout halves each)
  _Float16* pw10 = (_Float16*)take((size_t)32  * 64   * 2);
  _Float16* pw11 = (_Float16*)take((size_t)64  * 64   * 2);
  _Float16* pw12 = (_Float16*)take((size_t)64  * 128  * 2);
  _Float16* pw20 = (_Float16*)take((size_t)160 * 128  * 2);
  _Float16* pw21 = (_Float16*)take((size_t)128 * 128  * 2);
  _Float16* pw22 = (_Float16*)take((size_t)128 * 256  * 2);
  _Float16* pw30 = (_Float16*)take((size_t)288 * 256  * 2);
  _Float16* pw31 = (_Float16*)take((size_t)256 * 512  * 2);
  _Float16* pw32 = (_Float16*)take((size_t)512 * 1024 * 2);

  auto pack = [&](const float* W, int Cin, int Cout, _Float16* P) {
    int total = ((Cin + 31) & ~31) * Cout;
    pack_w_kernel<<<(total + 255) / 256, 256, 0, stream>>>(W, Cin, Cout, P);
  };
  pack(w10, 3,   64,   pw10);  pack(w11, 64,  64,   pw11);  pack(w12, 64,  128,  pw12);
  pack(w20, 131, 128,  pw20);  pack(w21, 128, 128,  pw21);  pack(w22, 128, 256,  pw22);
  pack(w30, 259, 256,  pw30);  pack(w31, 256, 512,  pw31);  pack(w32, 512, 1024, pw32);

  // ---- SA1 ----
  fps_kernel<<<BATCH, 256, 0, stream>>>(pc, NPTS, 512, xyz1);
  ballquery_kernel<<<(BATCH * 512 + 127) / 128, 128, 0, stream>>>(pc, NPTS, xyz1, BATCH * 512, 0.04f, idx1);
  sa1_kernel<<<BATCH * 512, 128, 0, stream>>>(pc, idx1, xyz1,
                                              pw10, g10, b10, pw11, g11, b11, pw12, g12, b12, feat1);
  // ---- SA2 ----
  fps_kernel<<<BATCH, 256, 0, stream>>>(xyz1, 512, 128, xyz2);
  ballquery_kernel<<<(BATCH * 128 + 127) / 128, 128, 0, stream>>>(xyz1, 512, xyz2, BATCH * 128, 0.16f, idx2);
  sa2_kernel<<<BATCH * 128, 128, 0, stream>>>(xyz1, idx2, xyz2, feat1,
                                              pw20, g20, b20, pw21, g21, b21, pw22, g22, b22, feat2);
  // ---- SA3 (global) ----
  build_a3_kernel<<<(BATCH * 128 * 288 + 255) / 256, 256, 0, stream>>>(xyz2, feat2, a3);
  gemm_relu_kernel<<<dim3(256 / 64, BATCH * 128 / 64), 128, 0, stream>>>(
      a3, 288, pw30, g30, b30, 259, 256, h31, nullptr, 256);
  gemm_relu_kernel<<<dim3(512 / 64, BATCH * 128 / 64), 128, 0, stream>>>(
      h31, 256, pw31, g31, b31, 256, 512, h32, nullptr, 512);
  gemm_relu_kernel<<<dim3(1024 / 64, BATCH * 128 / 64), 128, 0, stream>>>(
      h32, 512, pw32, g32, b32, 512, 1024, nullptr, h33, 1024);
  max3_kernel<<<dim3(BATCH, 4), 256, 0, stream>>>(h33, (float*)d_out);
}
